// DecoderBlock_69767448756346
// MI455X (gfx1250) — matrix operand order repaired and checked by reading, not compiled, not run on hardware
//
#include <hip/hip_runtime.h>
#include <math.h>

typedef __bf16 bf16_t;
typedef __attribute__((ext_vector_type(16))) __bf16 v16bf;
typedef __attribute__((ext_vector_type(8)))  float  v8f;
typedef __attribute__((ext_vector_type(4)))  int    v4i;

struct __align__(16) U4 { unsigned x, y, z, w; };
union AFrag { v16bf v; U4 u[2]; };

typedef __attribute__((address_space(1))) v4i v4i_g;   // global int4
typedef __attribute__((address_space(3))) v4i v4i_l;   // LDS int4

// ---- CDNA5 async global->LDS copy (16 B per lane), with graceful fallbacks ----
__device__ __forceinline__ void async_cp16(void* lds_dst, const void* gsrc) {
#if __has_builtin(__builtin_amdgcn_global_load_async_to_lds_b128)
  __builtin_amdgcn_global_load_async_to_lds_b128((v4i_g*)gsrc, (v4i_l*)lds_dst, 0, 0);
#elif defined(__gfx1250__)
  unsigned la = (unsigned)(unsigned long long)(v4i_l*)lds_dst;
  unsigned long long ga = (unsigned long long)gsrc;
  asm volatile("global_load_async_to_lds_b128 %0, %1, off" :: "v"(la), "v"(ga) : "memory");
#else
  *(U4*)lds_dst = *(const U4*)gsrc;
#endif
}

__device__ __forceinline__ void async_wait0() {
#if __has_builtin(__builtin_amdgcn_s_wait_asynccnt)
  __builtin_amdgcn_s_wait_asynccnt(0);
#elif defined(__gfx1250__)
  asm volatile("s_wait_asynccnt 0" ::: "memory");
#endif
}

__device__ __forceinline__ bf16_t f2bf(float f) {
  union { float f; unsigned u; } c; c.f = f;
  unsigned r = c.u + 0x7fffu + ((c.u >> 16) & 1u);   // round-to-nearest-even
  unsigned short h = (unsigned short)(r >> 16);
  bf16_t b; __builtin_memcpy(&b, &h, 2); return b;
}

// ---------------- elementwise / packing ----------------
__global__ __launch_bounds__(256) void cvt_bf16_kernel(const float* __restrict__ s,
                                                       bf16_t* __restrict__ d, int n) {
  int i = blockIdx.x * 256 + threadIdx.x;
  if (i < n) d[i] = f2bf(s[i]);
}

// dst (N x Kd, bf16, row-major) = transpose of src (Kd x N, f32, row-major)
__global__ __launch_bounds__(256) void pack_bt_kernel(const float* __restrict__ src,
                                                      bf16_t* __restrict__ dst,
                                                      int N, int Kd) {
  int i = blockIdx.x * 256 + threadIdx.x;
  if (i >= N * Kd) return;
  int n = i / Kd, k = i - n * Kd;
  dst[i] = f2bf(src[(size_t)k * N + n]);
}

// Pack up to 3 per-head weights (H=16, M=1024, DH=64) into Bt (Ncols x 1024):
// Bt[c][m] = W_{c/1024}[h=(c%1024)/64][m][d=c%64]
__global__ __launch_bounds__(256) void pack_qkv_w_kernel(const float* W0, const float* W1,
                                                         const float* W2,
                                                         bf16_t* __restrict__ dst, int total) {
  int i = blockIdx.x * 256 + threadIdx.x;
  if (i >= total) return;
  int c = i >> 10, m = i & 1023;
  int sub = c >> 10, cc = c & 1023, hh = cc >> 6, dd = cc & 63;
  const float* W = (sub == 0) ? W0 : (sub == 1 ? W1 : W2);
  dst[i] = f2bf(W[((size_t)hh * 1024 + m) * 64 + dd]);
}

// ---------------- WMMA GEMM: C(MxN) = A(MxKd) * Bt(NxKd)^T ----------------
struct GemmParams {
  const bf16_t* A; const bf16_t* Bt;
  int Ncols, Kd;
  int emode;                 // 0: f32 out, 1: bf16 out, 2: QKV per-head scatter
  float* outF; bf16_t* outB;
  const float* bias0; const float* bias1; const float* bias2;
  const float* resid; int relu; int colbase;
  bf16_t* Qo; bf16_t* Ko; bf16_t* Vto;   // per-head layouts; Vto is (d x keys)
};

__global__ __launch_bounds__(256) void gemm_bf16_wmma(GemmParams p) {
  __shared__ __align__(16) unsigned char smem[32768];  // 2 buffers x (As 8K | Bs 8K)
  const int tid  = threadIdx.x;
  const int lane = tid & 31, wave = tid >> 5;
  const int hlf  = lane >> 4, r = lane & 15;
  const int wr = wave & 1, wc = wave >> 1;             // 2 x 4 wave grid
  const int rowBase = blockIdx.y * 128;
  const int colBase = blockIdx.x * 128;
  const int Kd = p.Kd;

  v8f acc[4][2];
  const v8f vz = {0.f,0.f,0.f,0.f,0.f,0.f,0.f,0.f};
#pragma unroll
  for (int i = 0; i < 4; ++i)
#pragma unroll
    for (int j = 0; j < 2; ++j) acc[i][j] = vz;

  auto stage = [&](int buf, int k0) {
    unsigned char* As = smem + buf * 16384;
    unsigned char* Bs = As + 8192;
#pragma unroll
    for (int i = 0; i < 2; ++i) {
      int idx = tid + i * 256;                 // 512 chunks of 16B per 8K tile
      int row = idx >> 2, ch = idx & 3;
      async_cp16(As + idx * 16,
                 (const unsigned char*)(p.A + (size_t)(rowBase + row) * Kd + k0) + ch * 16);
      async_cp16(Bs + idx * 16,
                 (const unsigned char*)(p.Bt + (size_t)(colBase + row) * Kd + k0) + ch * 16);
    }
  };

  stage(0, 0);
  async_wait0();
  __syncthreads();

  for (int k0 = 0; k0 < Kd; k0 += 32) {
    const int cur = (k0 >> 5) & 1;
    if (k0 + 32 < Kd) stage(cur ^ 1, k0 + 32);   // async prefetch next tile

    const unsigned char* As = smem + cur * 16384;
    const unsigned char* Bs = As + 8192;
    AFrag a[4], b[2];
#pragma unroll
    for (int i = 0; i < 4; ++i) {
      const unsigned char* ba = As + (wr * 64 + i * 16 + r) * 64;
      a[i].u[0] = *(const U4*)(ba + hlf * 16);
      a[i].u[1] = *(const U4*)(ba + 32 + hlf * 16);
    }
#pragma unroll
    for (int j = 0; j < 2; ++j) {
      const unsigned char* bb = Bs + (wc * 32 + j * 16 + r) * 64;
      b[j].u[0] = *(const U4*)(bb + hlf * 16);
      b[j].u[1] = *(const U4*)(bb + 32 + hlf * 16);
    }
#pragma unroll
    for (int i = 0; i < 4; ++i)
#pragma unroll
      for (int j = 0; j < 2; ++j)
        acc[i][j] = __builtin_amdgcn_wmma_f32_16x16x32_bf16(
            false, a[i].v, false, b[j].v, (short)0, acc[i][j], false, false);

    async_wait0();      // next-tile async loads landed
    __syncthreads();    // (also flushes DS in fallback path)
  }

  // epilogue
#pragma unroll
  for (int i = 0; i < 4; ++i) {
#pragma unroll
    for (int j = 0; j < 2; ++j) {
      int row0 = rowBase + wr * 64 + i * 16 + hlf * 8;
      int col  = colBase + wc * 32 + j * 16 + r;
#pragma unroll
      for (int jj = 0; jj < 8; ++jj) {
        float v = acc[i][j][jj];
        int rr = row0 + jj;
        if (p.emode == 2) {
          int sub_l = col >> 10;
          const float* bp = (sub_l == 0) ? p.bias0 : (sub_l == 1 ? p.bias1 : p.bias2);
          if (bp) v += bp[col & 1023];
          int cg = p.colbase + col;
          int sub = cg >> 10, cc = cg & 1023, hh = cc >> 6, dd = cc & 63;
          int nn = rr >> 10, q = rr & 1023;
          size_t bh = (size_t)(nn * 16 + hh) << 16;          // * 1024*64
          if (sub == 0)      p.Qo [bh + (size_t)q * 64 + dd]  = f2bf(v);
          else if (sub == 1) p.Ko [bh + (size_t)q * 64 + dd]  = f2bf(v);
          else               p.Vto[bh + (size_t)dd * 1024 + q] = f2bf(v);  // transposed V
        } else {
          if (p.bias0) v += p.bias0[col];
          if (p.resid) v += p.resid[(size_t)rr * p.Ncols + col];
          if (p.relu)  v = fmaxf(v, 0.f);
          size_t o = (size_t)rr * p.Ncols + col;
          if (p.emode == 0) p.outF[o] = v;
          else              p.outB[o] = f2bf(v);
        }
      }
    }
  }
}

// ---------------- fused flash attention (online softmax) ----------------
__device__ __forceinline__ float redmax16(float v) {
#pragma unroll
  for (int o = 1; o < 16; o <<= 1) v = fmaxf(v, __shfl_xor(v, o, 32));
  return v;
}
__device__ __forceinline__ float redsum16(float v) {
#pragma unroll
  for (int o = 1; o < 16; o <<= 1) v += __shfl_xor(v, o, 32);
  return v;
}

// Q,K: per-batch (1024 x 64) row-major; Vt: per-batch (64 x 1024) row-major.
// Y: (N*1024 x 1024) bf16, heads concatenated (ready as next GEMM A operand).
__global__ __launch_bounds__(256) void attn_fused_kernel(const bf16_t* __restrict__ Qg,
                                                         const bf16_t* __restrict__ Kg,
                                                         const bf16_t* __restrict__ Vtg,
                                                         bf16_t* __restrict__ Yg,
                                                         int masked) {
  __shared__ __align__(16) unsigned char smem[16384];
  bf16_t* Ks = (bf16_t*)smem;             // 32 keys x 64 d
  bf16_t* Vs = (bf16_t*)(smem + 4096);    // 64 d x 32 keys
  unsigned char* Ps = smem + 8192;        // 8 waves x 1 KiB (wave-private P patch)

  const int tid  = threadIdx.x;
  const int lane = tid & 31, wave = tid >> 5;
  const int hlf  = lane >> 4, r = lane & 15;
  const int batch = blockIdx.y;
  const int nn = batch >> 4, hh = batch & 15;
  const int qbase = blockIdx.x * 128;

  const bf16_t* Qb = Qg  + (size_t)batch * 65536;
  const bf16_t* Kb = Kg  + (size_t)batch * 65536;
  const bf16_t* Vb = Vtg + (size_t)batch * 65536;

  AFrag aq[2];  // 16 queries x 64 d, two k-steps
  {
    const unsigned char* qp =
        (const unsigned char*)(Qb + (size_t)(qbase + wave * 16 + r) * 64);
#pragma unroll
    for (int s = 0; s < 2; ++s) {
      aq[s].u[0] = *(const U4*)(qp + s * 64 + hlf * 16);
      aq[s].u[1] = *(const U4*)(qp + s * 64 + 32 + hlf * 16);
    }
  }

  const v8f vz = {0.f,0.f,0.f,0.f,0.f,0.f,0.f,0.f};
  v8f accO[4]; accO[0] = vz; accO[1] = vz; accO[2] = vz; accO[3] = vz;
  float mrow[8], lrow[8];
#pragma unroll
  for (int jj = 0; jj < 8; ++jj) { mrow[jj] = -3.0e38f; lrow[jj] = 0.f; }

  const int qrow0 = qbase + wave * 16 + hlf * 8;
  const int tmax  = masked ? ((qbase >> 5) + 4) : 32;   // skip fully-masked key tiles

  for (int t = 0; t < tmax; ++t) {
    const int kb = t * 32;
    { int row = tid >> 3, ch = tid & 7;                 // K tile: 32 x (8x16B)
      async_cp16((unsigned char*)Ks + tid * 16,
                 (const unsigned char*)Kb + ((size_t)(kb + row) * 8 + ch) * 16); }
    { int row = tid >> 2, ch = tid & 3;                 // Vt tile: 64 x (4x16B)
      async_cp16((unsigned char*)Vs + tid * 16,
                 (const unsigned char*)Vb + ((size_t)row * 128 + t * 4 + ch) * 16); }
    async_wait0();
    __syncthreads();

    v8f sAcc[2]; sAcc[0] = vz; sAcc[1] = vz;            // S: 16q x 32keys
#pragma unroll
    for (int ct = 0; ct < 2; ++ct)
#pragma unroll
      for (int ks = 0; ks < 2; ++ks) {
        AFrag bk;
        const unsigned char* bb =
            (const unsigned char*)Ks + (ct * 16 + r) * 128 + ks * 64 + hlf * 16;
        bk.u[0] = *(const U4*)bb;
        bk.u[1] = *(const U4*)(bb + 32);
        sAcc[ct] = __builtin_amdgcn_wmma_f32_16x16x32_bf16(
            false, aq[ks].v, false, bk.v, (short)0, sAcc[ct], false, false);
      }

    float p0[8], p1[8];
#pragma unroll
    for (int jj = 0; jj < 8; ++jj) {
      float v0 = sAcc[0][jj] * 0.125f;                  // 1/sqrt(64)
      float v1 = sAcc[1][jj] * 0.125f;
      if (masked) {
        int q = qrow0 + jj;
        if (kb + r > q)      v0 = -1.0e9f;              // replace, like reference
        if (kb + 16 + r > q) v1 = -1.0e9f;
      }
      float mt = redmax16(fmaxf(v0, v1));
      float mn = fmaxf(mrow[jj], mt);
      float e0 = __expf(v0 - mn);
      float e1 = __expf(v1 - mn);
      float rs = redsum16(e0 + e1);
      float alpha = __expf(mrow[jj] - mn);
      lrow[jj] = lrow[jj] * alpha + rs;
      mrow[jj] = mn;
#pragma unroll
      for (int dt = 0; dt < 4; ++dt) accO[dt][jj] *= alpha;
      p0[jj] = e0; p1[jj] = e1;
    }

    // C-layout -> A-layout via wave-private LDS patch (16 x 32 bf16)
    bf16_t* pw = (bf16_t*)(Ps + wave * 1024);
#pragma unroll
    for (int jj = 0; jj < 8; ++jj) {
      pw[(hlf * 8 + jj) * 32 + r]      = f2bf(p0[jj]);
      pw[(hlf * 8 + jj) * 32 + 16 + r] = f2bf(p1[jj]);
    }
    asm volatile("s_wait_dscnt 0" ::: "memory");

    AFrag ap;
    {
      const unsigned char* pr = (const unsigned char*)pw + r * 64;
      ap.u[0] = *(const U4*)(pr + hlf * 16);
      ap.u[1] = *(const U4*)(pr + 32 + hlf * 16);
    }
#pragma unroll
    for (int dt = 0; dt < 4; ++dt) {
      AFrag bv;
      const unsigned char* bb =
          (const unsigned char*)Vs + (dt * 16 + r) * 64 + hlf * 16;
      bv.u[0] = *(const U4*)bb;
      bv.u[1] = *(const U4*)(bb + 32);
      accO[dt] = __builtin_amdgcn_wmma_f32_16x16x32_bf16(
          false, ap.v, false, bv.v, (short)0, accO[dt], false, false);
    }
    __syncthreads();
  }

#pragma unroll
  for (int jj = 0; jj < 8; ++jj) {
    float inv = 1.0f / lrow[jj];
    int q = qrow0 + jj;
    size_t rowoff = (size_t)(nn * 1024 + q) * 1024 + hh * 64;
#pragma unroll
    for (int dt = 0; dt < 4; ++dt)
      Yg[rowoff + dt * 16 + r] = f2bf(accO[dt][jj] * inv);
  }
}

// ---------------- layernorm: g*(x-mean)/(std+eps)+s ----------------
__global__ __launch_bounds__(256) void ln_kernel(const float* __restrict__ x,
                                                 const float* __restrict__ g,
                                                 const float* __restrict__ s,
                                                 float* __restrict__ outF,
                                                 bf16_t* __restrict__ outB) {
  __shared__ float red[8];
  const int tid = threadIdx.x;
  const size_t base = (size_t)blockIdx.x * 1024;
  float v[4];
#pragma unroll
  for (int i = 0; i < 4; ++i) v[i] = x[base + tid + i * 256];
  float loc = v[0] + v[1] + v[2] + v[3];
#pragma unroll
  for (int o = 16; o >= 1; o >>= 1) loc += __shfl_xor(loc, o, 32);
  if ((tid & 31) == 0) red[tid >> 5] = loc;
  __syncthreads();
  float tot = 0.f;
#pragma unroll
  for (int w = 0; w < 8; ++w) tot += red[w];
  const float mean = tot * (1.0f / 1024.0f);
  __syncthreads();
  float d[4];
#pragma unroll
  for (int i = 0; i < 4; ++i) d[i] = v[i] - mean;
  loc = d[0]*d[0] + d[1]*d[1] + d[2]*d[2] + d[3]*d[3];
#pragma unroll
  for (int o = 16; o >= 1; o >>= 1) loc += __shfl_xor(loc, o, 32);
  if ((tid & 31) == 0) red[tid >> 5] = loc;
  __syncthreads();
  tot = 0.f;
#pragma unroll
  for (int w = 0; w < 8; ++w) tot += red[w];
  const float inv = 1.0f / (sqrtf(tot * (1.0f / 1024.0f)) + 1e-10f);
#pragma unroll
  for (int i = 0; i < 4; ++i) {
    int c = tid + i * 256;
    float o = g[c] * d[i] * inv + s[c];
    outF[base + c] = o;
    if (outB) outB[base + c] = f2bf(o);
  }
}

// ---------------- host orchestration ----------------
extern "C" void kernel_launch(void* const* d_in, const int* in_sizes, int n_in,
                              void* d_out, int out_size, void* d_ws, size_t ws_size,
                              hipStream_t stream) {
  (void)in_sizes; (void)n_in; (void)out_size; (void)ws_size;
  const float* dec = (const float*)d_in[0];
  const float* enc = (const float*)d_in[1];
  // d_in[2] = mask (causal, computed analytically in-kernel)
  const float* Wq1 = (const float*)d_in[3];  const float* bq1 = (const float*)d_in[4];
  const float* Wk1 = (const float*)d_in[5];  const float* bk1 = (const float*)d_in[6];
  const float* Wv1 = (const float*)d_in[7];  const float* bv1 = (const float*)d_in[8];
  const float* Wp1 = (const float*)d_in[9];  const float* bp1 = (const float*)d_in[10];
  const float* Wq2 = (const float*)d_in[11]; const float* bq2 = (const float*)d_in[12];
  const float* Wk2 = (const float*)d_in[13]; const float* bk2 = (const float*)d_in[14];
  const float* Wv2 = (const float*)d_in[15]; const float* bv2 = (const float*)d_in[16];
  const float* Wp2 = (const float*)d_in[17]; const float* bp2 = (const float*)d_in[18];
  const float* W1  = (const float*)d_in[19]; const float* b1  = (const float*)d_in[20];
  const float* W2  = (const float*)d_in[21]; const float* b2  = (const float*)d_in[22];
  const float* g1  = (const float*)d_in[23]; const float* s1  = (const float*)d_in[24];
  const float* g2  = (const float*)d_in[25]; const float* s2  = (const float*)d_in[26];
  const float* g3  = (const float*)d_in[27]; const float* s3  = (const float*)d_in[28];
  float* out = (float*)d_out;

  char* w = (char*)d_ws;
  const size_t SZ_ACT_BF = (size_t)8192 * 1024 * 2;          // 16 MiB
  const size_t SZ_F32    = (size_t)8192 * 1024 * 4;          // 32 MiB
  const size_t SZ_WBUF   = (size_t)4096 * 1024 * 2;          // 8 MiB
  const size_t SZ_HEAD   = (size_t)128 * 1024 * 64 * 2;      // 16 MiB
  bf16_t* dec_bf = (bf16_t*)w; w += SZ_ACT_BF;
  bf16_t* enc_bf = (bf16_t*)w; w += SZ_ACT_BF;
  bf16_t* o1_bf  = (bf16_t*)w; w += SZ_ACT_BF;
  bf16_t* o2_bf  = (bf16_t*)w; w += SZ_ACT_BF;
  bf16_t* wbuf   = (bf16_t*)w; w += SZ_WBUF;
  bf16_t* Qb     = (bf16_t*)w; w += SZ_HEAD;
  bf16_t* Kb     = (bf16_t*)w; w += SZ_HEAD;
  bf16_t* Vtb    = (bf16_t*)w; w += SZ_HEAD;
  bf16_t* Yb     = (bf16_t*)w; w += SZ_HEAD;
  bf16_t* hid    = Qb;   // 64 MiB FFN hidden aliases Qb..Yb (dead by then)
  float*  sumf   = (float*)w; w += SZ_F32;
  float*  o1f    = (float*)w; w += SZ_F32;
  float*  o2f    = (float*)w; w += SZ_F32;

  const dim3 blk(256);
  cvt_bf16_kernel<<<32768, blk, 0, stream>>>(dec, dec_bf, 8192 * 1024);
  cvt_bf16_kernel<<<32768, blk, 0, stream>>>(enc, enc_bf, 8192 * 1024);

  // ---- masked self-attention ----
  pack_qkv_w_kernel<<<(3072 * 1024) / 256, blk, 0, stream>>>(Wq1, Wk1, Wv1, wbuf, 3072 * 1024);
  { GemmParams p = {}; p.A = dec_bf; p.Bt = wbuf; p.Ncols = 3072; p.Kd = 1024; p.emode = 2;
    p.bias0 = bq1; p.bias1 = bk1; p.bias2 = bv1; p.colbase = 0;
    p.Qo = Qb; p.Ko = Kb; p.Vto = Vtb;
    gemm_bf16_wmma<<<dim3(24, 64), blk, 0, stream>>>(p); }
  attn_fused_kernel<<<dim3(8, 128), blk, 0, stream>>>(Qb, Kb, Vtb, Yb, 1);
  pack_bt_kernel<<<(1024 * 1024) / 256, blk, 0, stream>>>(Wp1, wbuf, 1024, 1024);
  { GemmParams p = {}; p.A = Yb; p.Bt = wbuf; p.Ncols = 1024; p.Kd = 1024; p.emode = 0;
    p.outF = sumf; p.bias0 = bp1; p.resid = dec;
    gemm_bf16_wmma<<<dim3(8, 64), blk, 0, stream>>>(p); }
  ln_kernel<<<8192, blk, 0, stream>>>(sumf, g1, s1, o1f, o1_bf);

  // ---- cross-attention ----
  pack_qkv_w_kernel<<<(1024 * 1024) / 256, blk, 0, stream>>>(Wq2, nullptr, nullptr, wbuf, 1024 * 1024);
  { GemmParams p = {}; p.A = o1_bf; p.Bt = wbuf; p.Ncols = 1024; p.Kd = 1024; p.emode = 2;
    p.bias0 = bq2; p.colbase = 0; p.Qo = Qb; p.Ko = Kb; p.Vto = Vtb;
    gemm_bf16_wmma<<<dim3(8, 64), blk, 0, stream>>>(p); }
  pack_qkv_w_kernel<<<(2048 * 1024) / 256, blk, 0, stream>>>(Wk2, Wv2, nullptr, wbuf, 2048 * 1024);
  { GemmParams p = {}; p.A = enc_bf; p.Bt = wbuf; p.Ncols = 2048; p.Kd = 1024; p.emode = 2;
    p.bias0 = bk2; p.bias1 = bv2; p.colbase = 1024; p.Qo = Qb; p.Ko = Kb; p.Vto = Vtb;
    gemm_bf16_wmma<<<dim3(16, 64), blk, 0, stream>>>(p); }
  attn_fused_kernel<<<dim3(8, 128), blk, 0, stream>>>(Qb, Kb, Vtb, Yb, 0);
  pack_bt_kernel<<<(1024 * 1024) / 256, blk, 0, stream>>>(Wp2, wbuf, 1024, 1024);
  { GemmParams p = {}; p.A = Yb; p.Bt = wbuf; p.Ncols = 1024; p.Kd = 1024; p.emode = 0;
    p.outF = sumf; p.bias0 = bp2; p.resid = o1f;
    gemm_bf16_wmma<<<dim3(8, 64), blk, 0, stream>>>(p); }
  ln_kernel<<<8192, blk, 0, stream>>>(sumf, g2, s2, o2f, o2_bf);

  // ---- FFN ----
  pack_bt_kernel<<<(4096 * 1024) / 256, blk, 0, stream>>>(W1, wbuf, 4096, 1024);
  { GemmParams p = {}; p.A = o2_bf; p.Bt = wbuf; p.Ncols = 4096; p.Kd = 1024; p.emode = 1;
    p.outB = hid; p.bias0 = b1; p.relu = 1;
    gemm_bf16_wmma<<<dim3(32, 64), blk, 0, stream>>>(p); }
  pack_bt_kernel<<<(4096 * 1024) / 256, blk, 0, stream>>>(W2, wbuf, 1024, 4096);
  { GemmParams p = {}; p.A = hid; p.Bt = wbuf; p.Ncols = 1024; p.Kd = 4096; p.emode = 0;
    p.outF = sumf; p.bias0 = b2; p.resid = o2f;
    gemm_bf16_wmma<<<dim3(8, 64), blk, 0, stream>>>(p); }
  ln_kernel<<<8192, blk, 0, stream>>>(sumf, g3, s3, out, nullptr);
}
